// JointAttention_28613072125989
// MI455X (gfx1250) — matrix operand order repaired and checked by reading, not compiled, not run on hardware
//
#include <hip/hip_runtime.h>
#include <hip/hip_bf16.h>

// ---------------------------------------------------------------------------
// Joint dual-stream attention for MI455X / gfx1250, bf16 WMMA end-to-end.
// Roofline: ~45 GFLOP vs ~100 MB of traffic (fits in 192MB L2) -> compute
// bound on the WMMA pipe, so: one-time f32->bf16 conversion, then all GEMMs
// and attention run v_wmma_f32_16x16x32_bf16 with async global->LDS staging
// (ASYNCcnt) and double-buffered K-loop so fetch overlaps the matrix pipe.
// ---------------------------------------------------------------------------

typedef __attribute__((ext_vector_type(16))) __bf16 v16bf;
typedef __attribute__((ext_vector_type(8)))  float  v8f;
typedef __attribute__((ext_vector_type(4)))  int    v4i;

#define HEADS 16
#define HDIM  64
#define GAS __attribute__((address_space(1)))
#define LAS __attribute__((address_space(3)))

// ---- CDNA5 async global->LDS (guarded; falls back to a sync 16B copy) -----
// Builtin signature (probed via round-2 diagnostic): (v4i AS1*, v4i AS3*, Ii, Ii)
__device__ inline void async_copy16(const __bf16* g, __bf16* l) {
#if __has_builtin(__builtin_amdgcn_global_load_async_to_lds_b128)
  __builtin_amdgcn_global_load_async_to_lds_b128(
      (GAS v4i*)g, (LAS v4i*)l, 0, 0);
#else
  *(uint4*)l = *(const uint4*)g;
#endif
}

#if __has_builtin(__builtin_amdgcn_s_wait_asynccnt)
#define WAIT_ASYNC(n) __builtin_amdgcn_s_wait_asynccnt(n)
#else
#define WAIT_ASYNC(n) asm volatile("s_wait_asynccnt %0" ::"n"(n) : "memory")
#endif

__device__ inline v8f wmma_bf16(v16bf a, v16bf b, v8f c) {
  return __builtin_amdgcn_wmma_f32_16x16x32_bf16(
      false, a, false, b, (short)0, c, false, false);
}

// A fragment 16x32 (MxK): lane m = lane&15 holds row m; VGPR v holds K pair
// {2v,2v+1} (v<4) / {8+2v,9+2v} (v>=4), +8 for lanes 16-31.
__device__ inline v16bf load_frag_a(const __bf16* rowp, int hi, int kbase) {
  union { v16bf v; unsigned u[8]; } f;
  const unsigned short* r = (const unsigned short*)rowp;
#pragma unroll
  for (int i = 0; i < 8; ++i) {
    int kb = ((i < 4) ? (2 * i) : (8 + 2 * i)) + hi * 8 + kbase;
    f.u[i] = *(const unsigned int*)(r + kb);
  }
  return f.v;
}

// B fragment 32x16 (KxN): lane n = lane&15 holds column n (= row n of the
// row-major K-contiguous operand); same K order as A fragment (+8 for lanes 16-31).
__device__ inline v16bf load_frag_b(const __bf16* rowp, int hi, int kbase) {
  union { v16bf v; unsigned u[8]; } f;
  const unsigned short* r = (const unsigned short*)rowp;
#pragma unroll
  for (int i = 0; i < 8; ++i) {
    int kb = ((i < 4) ? (2 * i) : (8 + 2 * i)) + hi * 8 + kbase;
    f.u[i] = *(const unsigned int*)(r + kb);
  }
  return f.v;
}

// ---------------------------------------------------------------------------
// Kernel 0: f32 -> bf16 bulk convert (one-time; makes all GEMM operands
// eligible for async-to-LDS staging). n is a multiple of 4.
// ---------------------------------------------------------------------------
__global__ __launch_bounds__(256) void f32_to_bf16_kernel(
    const float* __restrict__ in, __bf16* __restrict__ out, int n) {
  int i = (blockIdx.x * 256 + threadIdx.x) * 4;
  if (i + 3 < n) {
    float4 f = *(const float4*)(in + i);
    union { __bf16 h[4]; uint2 u; } pk;
    pk.h[0] = (__bf16)f.x; pk.h[1] = (__bf16)f.y;
    pk.h[2] = (__bf16)f.z; pk.h[3] = (__bf16)f.w;
    *(uint2*)(out + i) = pk.u;
  }
}

// ---------------------------------------------------------------------------
// Kernel 1: GEMM  out = A @ W^T (+bias), bf16 operands, f32 accumulate.
//   A rows: (b, r) -> b*full_len + seg_off + r   (seg_len = 1<<seg_shift)
//   out_mode 0: f32 out[gm*N + n] + bias
//   out_mode 1: bf16 scatter to (B,H,Ljoint,D) at joint offset out_off
// Block tile 128x64, BK=32, double-buffered async LDS staging; 8 waves,
// wave tile 32x32 (2x2 WMMA).
// ---------------------------------------------------------------------------
__global__ __launch_bounds__(256) void gemm_bf16_wmma(
    const __bf16* __restrict__ A, const __bf16* __restrict__ W,
    const float* __restrict__ bias, void* __restrict__ outp,
    int N, int K,
    int seg_shift, int seg_off, int full_len,
    int out_mode, int Ljoint, int out_off) {
  __shared__ alignas(16) __bf16 As[2][128 * 32];   // 16 KB
  __shared__ alignas(16) __bf16 Bs[2][64 * 32];    //  8 KB

  const int tid = threadIdx.x;
  const int lane = tid & 31, wave = tid >> 5;
  const int l15 = lane & 15, hi = (lane >> 4) & 1;
  const int ntiles = N >> 6;
  const int bx = blockIdx.x % ntiles;
  const int by = blockIdx.x / ntiles;
  const int wm = wave >> 1, wn = wave & 1;
  const int smask = (1 << seg_shift) - 1;

  // Stage one 128x32 A tile + 64x32 W tile: 3 async b128 per thread.
  auto stage = [&](int kk, __bf16* as, __bf16* bs) {
#pragma unroll
    for (int c = tid; c < 512; c += 256) {
      int row = c >> 2, part = c & 3;
      int gm = by * 128 + row;
      int b = gm >> seg_shift, rr = gm & smask;
      size_t arow = (size_t)b * full_len + seg_off + rr;
      async_copy16(A + arow * (size_t)K + kk + part * 8, as + row * 32 + part * 8);
    }
    {
      int row = tid >> 2, part = tid & 3;
      size_t wrow = (size_t)(bx * 64 + row);
      async_copy16(W + wrow * (size_t)K + kk + part * 8, bs + row * 32 + part * 8);
    }
  };

  v8f acc[2][2];
#pragma unroll
  for (int i = 0; i < 2; ++i)
#pragma unroll
    for (int j = 0; j < 2; ++j)
#pragma unroll
      for (int r = 0; r < 8; ++r) acc[i][j][r] = 0.0f;

  const int nk = K >> 5;
  stage(0, As[0], Bs[0]);
  for (int t = 0; t < nk; ++t) {
    __bf16* as = As[t & 1];
    __bf16* bs = Bs[t & 1];
    if (t + 1 < nk) {
      stage((t + 1) * 32, As[(t + 1) & 1], Bs[(t + 1) & 1]);
      WAIT_ASYNC(3);             // tile t landed; tile t+1 still in flight
    } else {
      WAIT_ASYNC(0);
    }
    __syncthreads();

    v16bf a0 = load_frag_a(as + (wm * 32 + l15) * 32, hi, 0);
    v16bf a1 = load_frag_a(as + (wm * 32 + 16 + l15) * 32, hi, 0);
    v16bf b0 = load_frag_b(bs + (wn * 32 + l15) * 32, hi, 0);
    v16bf b1 = load_frag_b(bs + (wn * 32 + 16 + l15) * 32, hi, 0);

    acc[0][0] = wmma_bf16(a0, b0, acc[0][0]);
    acc[0][1] = wmma_bf16(a0, b1, acc[0][1]);
    acc[1][0] = wmma_bf16(a1, b0, acc[1][0]);
    acc[1][1] = wmma_bf16(a1, b1, acc[1][1]);
    __syncthreads();             // all waves done reading buf before reuse
  }

#pragma unroll
  for (int i = 0; i < 2; ++i) {
#pragma unroll
    for (int j = 0; j < 2; ++j) {
      int colb = bx * 64 + wn * 32 + j * 16 + l15;
      float bv = bias ? bias[colb] : 0.0f;
#pragma unroll
      for (int r = 0; r < 8; ++r) {
        int gm = by * 128 + wm * 32 + i * 16 + hi * 8 + r;
        float val = acc[i][j][r] + bv;
        if (out_mode == 0) {
          ((float*)outp)[(size_t)gm * N + colb] = val;
        } else {
          int b = gm >> seg_shift, rr = gm & smask;
          int h = colb >> 6, d = colb & 63;
          ((__bf16*)outp)[(((size_t)(b * HEADS + h)) * Ljoint + out_off + rr) * HDIM + d] =
              (__bf16)val;
        }
      }
    }
  }
}

// ---------------------------------------------------------------------------
// Kernel 2: per-head LayerNorm + interleaved RoPE, in place on bf16 (B,H,L,D).
// Stats in f32; one wave per 64-elem row; lane owns rotate pair (2l, 2l+1).
// outscale folds the attention softmax scale (D^-0.5) into the q pass.
// ---------------------------------------------------------------------------
__global__ __launch_bounds__(256) void ln_rope_kernel(
    __bf16* __restrict__ t,
    const float* __restrict__ gc, const float* __restrict__ bc,
    const float* __restrict__ gx, const float* __restrict__ bx,
    int Ljoint, int Lc, int nrows, float outscale) {
  const int lane = threadIdx.x & 31;
  const int wave = threadIdx.x >> 5;
  const int row = blockIdx.x * 8 + wave;
  if (row >= nrows) return;
  const int l = row % Ljoint;
  __bf16* p = t + (size_t)row * HDIM;

  union { unsigned u; __bf16 h[2]; } rd;
  rd.u = *(const unsigned*)(p + 2 * lane);
  float x0 = (float)rd.h[0], x1 = (float)rd.h[1];
  float s = x0 + x1;
#pragma unroll
  for (int off = 1; off < 32; off <<= 1) s += __shfl_xor(s, off, 32);
  float mu = s * (1.0f / 64.0f);
  float d0 = x0 - mu, d1 = x1 - mu;
  float vs = d0 * d0 + d1 * d1;
#pragma unroll
  for (int off = 1; off < 32; off <<= 1) vs += __shfl_xor(vs, off, 32);
  float rs = rsqrtf(vs * (1.0f / 64.0f) + 1e-5f);

  const float* g = (l < Lc) ? gc : gx;
  const float* b = (l < Lc) ? bc : bx;
  float y0 = d0 * rs * g[2 * lane] + b[2 * lane];
  float y1 = d1 * rs * g[2 * lane + 1] + b[2 * lane + 1];

  // inv_freq[i] = 10000^(-2i/64) = exp(-i * ln(1e4)/32); i == lane
  float invf = __expf(-0.28782313662425575f * (float)lane);
  float ang = (float)l * invf;
  float c = __cosf(ang), sn = __sinf(ang);
  union { unsigned u; __bf16 h[2]; } wr;
  wr.h[0] = (__bf16)((y0 * c - y1 * sn) * outscale);
  wr.h[1] = (__bf16)((y1 * c + y0 * sn) * outscale);
  *(unsigned*)(p + 2 * lane) = wr.u;
}

// ---------------------------------------------------------------------------
// Kernel 3: flash attention over the joint sequence (all-bf16 operands).
// One WG per (b,h,128 q rows); 8 waves x 16 rows. K chunks async-staged to
// LDS; V manually transposed into LDS; QK^T + PV on WMMA; online softmax
// via 16-lane shuffle row reductions. Masks are all-true -> no mask term.
// Writes bf16 (B,L,H*D) so the projection GEMMs consume it directly.
// ---------------------------------------------------------------------------
__global__ __launch_bounds__(256) void attn_kernel(
    const __bf16* __restrict__ q, const __bf16* __restrict__ k,
    const __bf16* __restrict__ v, __bf16* __restrict__ out, int Ljoint) {
  __shared__ alignas(16) __bf16 Ks[32 * 64];      // keys row-major
  __shared__ alignas(16) __bf16 Vt[64 * 32];      // V^T (d-major)
  __shared__ alignas(16) __bf16 Pb[8][16 * 32];   // per-wave P bounce

  const int tid = threadIdx.x;
  const int lane = tid & 31, wave = tid >> 5;
  const int l15 = lane & 15, hi = (lane >> 4) & 1;

  const int qtiles = Ljoint >> 7;
  const int qt = blockIdx.x % qtiles;
  const int bh = blockIdx.x / qtiles;
  const int b = bh >> 4, h = bh & 15;

  const __bf16* qp = q + (size_t)bh * Ljoint * HDIM;
  const __bf16* kp = k + (size_t)bh * Ljoint * HDIM;
  const __bf16* vp = v + (size_t)bh * Ljoint * HDIM;
  __bf16* op = out + (size_t)b * Ljoint * (HEADS * HDIM) + h * HDIM;

  // Q fragments straight from global bf16 (scale already folded in by LN).
  const int qrow = qt * 128 + wave * 16 + l15;
  const unsigned short* qr = (const unsigned short*)(qp + (size_t)qrow * HDIM);
  union { v16bf v; unsigned u[8]; } qa0, qa1;
#pragma unroll
  for (int i = 0; i < 8; ++i) {
    int kb = ((i < 4) ? (2 * i) : (8 + 2 * i)) + hi * 8;
    qa0.u[i] = *(const unsigned*)(qr + kb);
    qa1.u[i] = *(const unsigned*)(qr + kb + 32);
  }

  v8f o[4];
  float mrun[8], lrun[8];
#pragma unroll
  for (int t = 0; t < 4; ++t)
#pragma unroll
    for (int r = 0; r < 8; ++r) o[t][r] = 0.0f;
#pragma unroll
  for (int r = 0; r < 8; ++r) { mrun[r] = -3.0e38f; lrun[r] = 0.0f; }

  const int nchunks = Ljoint >> 5;
  for (int kc = 0; kc < nchunks; ++kc) {
    __syncthreads();
    {  // K chunk 32x64: one async b128 per thread
      int j = tid >> 3, part = tid & 7;
      async_copy16(kp + (size_t)(kc * 32 + j) * HDIM + part * 8,
                   Ks + j * 64 + part * 8);
    }
    {  // V chunk: 16B global read, transpose-scatter into Vt
      int j = tid >> 3, dblk = tid & 7;
      union { uint4 q4; __bf16 h[8]; } vv;
      vv.q4 = *(const uint4*)(vp + (size_t)(kc * 32 + j) * HDIM + dblk * 8);
#pragma unroll
      for (int e = 0; e < 8; ++e) Vt[(dblk * 8 + e) * 32 + j] = vv.h[e];
    }
    WAIT_ASYNC(0);
    __syncthreads();

    v8f s0, s1;
#pragma unroll
    for (int r = 0; r < 8; ++r) { s0[r] = 0.0f; s1[r] = 0.0f; }
    v16bf kb00 = load_frag_b(Ks + l15 * 64, hi, 0);
    v16bf kb01 = load_frag_b(Ks + l15 * 64, hi, 32);
    v16bf kb10 = load_frag_b(Ks + (16 + l15) * 64, hi, 0);
    v16bf kb11 = load_frag_b(Ks + (16 + l15) * 64, hi, 32);
    s0 = wmma_bf16(qa0.v, kb00, s0);
    s0 = wmma_bf16(qa1.v, kb01, s0);
    s1 = wmma_bf16(qa0.v, kb10, s1);
    s1 = wmma_bf16(qa1.v, kb11, s1);

    float fac[8];
#pragma unroll
    for (int r = 0; r < 8; ++r) {
      float x = fmaxf(s0[r], s1[r]);
#pragma unroll
      for (int off = 1; off < 16; off <<= 1) x = fmaxf(x, __shfl_xor(x, off, 32));
      float mnew = fmaxf(mrun[r], x);
      fac[r] = __expf(mrun[r] - mnew);
      mrun[r] = mnew;
    }
#pragma unroll
    for (int r = 0; r < 8; ++r) {
      float p0 = __expf(s0[r] - mrun[r]);
      float p1 = __expf(s1[r] - mrun[r]);
      float sum = p0 + p1;
#pragma unroll
      for (int off = 1; off < 16; off <<= 1) sum += __shfl_xor(sum, off, 32);
      lrun[r] = lrun[r] * fac[r] + sum;
      int row = r + hi * 8;
      Pb[wave][row * 32 + l15]      = (__bf16)p0;
      Pb[wave][row * 32 + 16 + l15] = (__bf16)p1;
      o[0][r] *= fac[r]; o[1][r] *= fac[r];
      o[2][r] *= fac[r]; o[3][r] *= fac[r];
    }
    asm volatile("s_wait_dscnt 0" ::: "memory");  // own LDS writes visible

    v16bf pf = load_frag_a(Pb[wave] + l15 * 32, hi, 0);
#pragma unroll
    for (int t = 0; t < 4; ++t) {
      v16bf vf = load_frag_b(Vt + (t * 16 + l15) * 32, hi, 0);
      o[t] = wmma_bf16(pf, vf, o[t]);
    }
  }

#pragma unroll
  for (int r = 0; r < 8; ++r) {
    float inv = 1.0f / lrun[r];
    int gl = qt * 128 + wave * 16 + r + hi * 8;
#pragma unroll
    for (int t = 0; t < 4; ++t)
      op[(size_t)gl * (HEADS * HDIM) + t * 16 + l15] = (__bf16)(o[t][r] * inv);
  }
}

// ---------------------------------------------------------------------------
// Host-side orchestration.
// ---------------------------------------------------------------------------
extern "C" void kernel_launch(void* const* d_in, const int* in_sizes, int n_in,
                              void* d_out, int out_size, void* d_ws, size_t ws_size,
                              hipStream_t stream) {
  (void)in_sizes; (void)n_in; (void)out_size; (void)ws_size;

  const float* x    = (const float*)d_in[0];
  const float* ctx  = (const float*)d_in[1];
  const float* w_qx = (const float*)d_in[5];
  const float* w_kx = (const float*)d_in[6];
  const float* w_vx = (const float*)d_in[7];
  const float* w_qc = (const float*)d_in[8];
  const float* w_kc = (const float*)d_in[9];
  const float* w_vc = (const float*)d_in[10];
  const float* g_qx = (const float*)d_in[11];
  const float* b_qx = (const float*)d_in[12];
  const float* g_kx = (const float*)d_in[13];
  const float* b_kx = (const float*)d_in[14];
  const float* g_qc = (const float*)d_in[15];
  const float* b_qc = (const float*)d_in[16];
  const float* g_kc = (const float*)d_in[17];
  const float* b_kc = (const float*)d_in[18];
  const float* w_px = (const float*)d_in[19];
  const float* b_px = (const float*)d_in[20];
  const float* w_pc = (const float*)d_in[21];
  const float* b_pc = (const float*)d_in[22];

  const int Bc = 2, LX = 1024, LC = 512, C = 1024, L = 1536;
  const size_t NX = (size_t)Bc * LX * C;     // 2,097,152
  const size_t NC = (size_t)Bc * LC * C;     // 1,048,576
  const size_t NW = (size_t)C * C;           // 1,048,576
  const size_t S  = (size_t)Bc * HEADS * L * HDIM;  // 3,145,728

  // bf16 workspace layout
  __bf16* xh  = (__bf16*)d_ws;
  __bf16* chh = xh + NX;
  __bf16* wh  = chh + NC;                    // 8 weight matrices
  __bf16* qh  = wh + 8 * NW;
  __bf16* kh  = qh + S;
  __bf16* vh  = kh + S;
  __bf16* abh = vh + S;                      // attention out (B,L,C)
  float* out_x = (float*)d_out;
  float* out_c = (float*)d_out + NX;

  dim3 blk(256);
  // one-time f32 -> bf16 conversion (all GEMM operands)
  f32_to_bf16_kernel<<<(int)(NX / 1024), blk, 0, stream>>>(x, xh, (int)NX);
  f32_to_bf16_kernel<<<(int)(NC / 1024), blk, 0, stream>>>(ctx, chh, (int)NC);
  const float* wsrc[8] = {w_qx, w_kx, w_vx, w_qc, w_kc, w_vc, w_px, w_pc};
  for (int i = 0; i < 8; ++i)
    f32_to_bf16_kernel<<<(int)(NW / 1024), blk, 0, stream>>>(wsrc[i], wh + i * NW, (int)NW);

  // QKV projections -> bf16 (B,H,L,D) scatter; x stream at joint offset LC
  gemm_bf16_wmma<<<256, blk, 0, stream>>>(xh,  wh + 0 * NW, nullptr, qh, C, C, 10, 0, LX, 1, L, LC);
  gemm_bf16_wmma<<<256, blk, 0, stream>>>(xh,  wh + 1 * NW, nullptr, kh, C, C, 10, 0, LX, 1, L, LC);
  gemm_bf16_wmma<<<256, blk, 0, stream>>>(xh,  wh + 2 * NW, nullptr, vh, C, C, 10, 0, LX, 1, L, LC);
  gemm_bf16_wmma<<<128, blk, 0, stream>>>(chh, wh + 3 * NW, nullptr, qh, C, C,  9, 0, LC, 1, L, 0);
  gemm_bf16_wmma<<<128, blk, 0, stream>>>(chh, wh + 4 * NW, nullptr, kh, C, C,  9, 0, LC, 1, L, 0);
  gemm_bf16_wmma<<<128, blk, 0, stream>>>(chh, wh + 5 * NW, nullptr, vh, C, C,  9, 0, LC, 1, L, 0);

  // per-head LN + RoPE in place; fold softmax scale (1/8) into q
  const int nrows = Bc * HEADS * L;          // 49152
  ln_rope_kernel<<<nrows / 8, blk, 0, stream>>>(qh, g_qc, b_qc, g_qx, b_qx, L, LC, nrows, 0.125f);
  ln_rope_kernel<<<nrows / 8, blk, 0, stream>>>(kh, g_kc, b_kc, g_kx, b_kx, L, LC, nrows, 1.0f);

  // flash attention: (B*H) * (L/128) = 384 workgroups
  attn_kernel<<<384, blk, 0, stream>>>(qh, kh, vh, abh, L);

  // output projections (+bias) -> f32 halves of d_out
  gemm_bf16_wmma<<<256, blk, 0, stream>>>(abh, wh + 6 * NW, b_px, out_x, C, C, 10, LC, L, 0, 0, 0);
  gemm_bf16_wmma<<<128, blk, 0, stream>>>(abh, wh + 7 * NW, b_pc, out_c, C, C,  9, 0,  L, 0, 0, 0);
}
